// GNN_Final_VN_Model_89094801588810
// MI455X (gfx1250) — matrix operand order repaired and checked by reading, not compiled, not run on hardware
//
#include <hip/hip_runtime.h>
#include <hip/hip_bf16.h>
#include <stdint.h>

#define N_NODES 50000
#define N_EDGES 400000
#define E2      (N_EDGES + N_NODES)   // edges + self-loops = 450000
#define DIM     128
#define EDIM    64

typedef __attribute__((ext_vector_type(16))) __bf16 v16bf;
typedef __attribute__((ext_vector_type(8)))  float  v8f;

union BF16Frag { v16bf v; uint4 q[2]; unsigned short s[16]; };

__device__ __forceinline__ unsigned short f2bf(float f) {
    unsigned u = __float_as_uint(f);
    u += 0x7FFFu + ((u >> 16) & 1u);           // round-to-nearest-even
    return (unsigned short)(u >> 16);
}

// ---------------------------------------------------------------------------
// Self-loop attr: deg[dst] += 1 ; loop_sum[dst][c] += edge_attr[e][c]
// ---------------------------------------------------------------------------
__global__ void deg_loop_kernel(const int* __restrict__ ei, const float* __restrict__ ea,
                                float* __restrict__ deg, float* __restrict__ loop_attr) {
    int idx = blockIdx.x * 256 + threadIdx.x;
    if (idx >= N_EDGES * EDIM) return;
    int e = idx >> 6, c = idx & 63;
    int dst = ei[N_EDGES + e];
    atomicAdd(&loop_attr[(size_t)dst * EDIM + c], ea[idx]);
    if (c == 0) atomicAdd(&deg[dst], 1.0f);
}

__global__ void loop_norm_kernel(const float* __restrict__ deg, float* __restrict__ loop_attr) {
    int idx = blockIdx.x * 256 + threadIdx.x;
    if (idx >= N_NODES * EDIM) return;
    int node = idx >> 6;
    loop_attr[idx] /= fmaxf(deg[node], 1.0f);
}

// ---------------------------------------------------------------------------
// wae[l][k] = sum_j We_l[k][j] * ae_l[j]   (3 layers x 64)
// ---------------------------------------------------------------------------
__global__ void wae_kernel(const float* We0, const float* ae0,
                           const float* We1, const float* ae1,
                           const float* We2, const float* ae2,
                           float* __restrict__ wae) {
    int t = threadIdx.x;
    if (t >= 3 * 64) return;
    int layer = t >> 6, k = t & 63;
    const float* We = (layer == 0) ? We0 : (layer == 1) ? We1 : We2;
    const float* ae = (layer == 0) ? ae0 : (layer == 1) ? ae1 : ae2;
    float s = 0.f;
    for (int j = 0; j < DIM; ++j) s += We[k * DIM + j] * ae[j];
    wae[t] = s;
}

// One pass over edge features: ae3[l][e] = dot(row(e), wae[l]) for all 3 layers.
__global__ void edge_ae_kernel(const float* __restrict__ ea, const float* __restrict__ loop_attr,
                               const float* __restrict__ wae, float* __restrict__ ae3) {
    int gid = blockIdx.x * 256 + threadIdx.x;
    int e = gid >> 4, s = gid & 15;
    if (e >= E2) return;
    const float* row = (e < N_EDGES) ? (ea + (size_t)e * EDIM)
                                     : (loop_attr + (size_t)(e - N_EDGES) * EDIM);
    float d0 = 0.f, d1 = 0.f, d2 = 0.f;
    for (int c = s; c < EDIM; c += 16) {
        float v = row[c];
        d0 += v * wae[c]; d1 += v * wae[64 + c]; d2 += v * wae[128 + c];
    }
    for (int ofs = 8; ofs; ofs >>= 1) {
        d0 += __shfl_xor(d0, ofs, 32);
        d1 += __shfl_xor(d1, ofs, 32);
        d2 += __shfl_xor(d2, ofs, 32);
    }
    if (s == 0) { ae3[e] = d0; ae3[E2 + e] = d1; ae3[2 * E2 + e] = d2; }
}

// ---------------------------------------------------------------------------
// WMMA GEMM: D[M,128] = A[M,128] @ W[128,128] (+bias), bf16 in, f32 acc.
// Block = 256 thr (8 waves) per 128-row strip. Wave j owns N-tile cols 16j,
// hoists its 4 B-fragments into registers once, then sweeps 8 M-tiles
// (32 v_wmma per wave). W staged once per 128 rows -> 8x less L2 traffic.
// ---------------------------------------------------------------------------
#define LROW 136                                   // padded LDS row: 272 B = 17 uint4
#define GEMM_BLOCKS ((N_NODES / 16 + 7) / 8)       // 391

__global__ void gemm_wmma_kernel(const float* __restrict__ A, const float* __restrict__ W,
                                 const float* __restrict__ bias, float* __restrict__ D) {
    __shared__ __align__(16) unsigned short Wt[DIM * LROW];   // W transposed [n][k], bf16
    __shared__ __align__(16) unsigned short As[128 * LROW];   // A strip [r][k],  bf16
    const int tid = threadIdx.x;
    const int row0 = blockIdx.x * 128;

    for (int idx = tid; idx < DIM * DIM; idx += 256) {        // coalesced W read, transpose to LDS
        int k = idx >> 7, n = idx & 127;
        Wt[n * LROW + k] = f2bf(W[idx]);
    }
    for (int idx = tid; idx < 128 * DIM; idx += 256) {        // stage A strip as bf16 (zero-pad tail)
        int r = idx >> 7, c = idx & 127;
        int row = row0 + r;
        As[r * LROW + c] = (row < N_NODES) ? f2bf(A[(size_t)row * DIM + c]) : (unsigned short)0;
    }
    __syncthreads();

    const int wave = tid >> 5, lane = tid & 31;
    const int m = lane & 15, hi = lane >> 4;
    const int n0 = wave * 16;
    const uint4* AsQ = (const uint4*)As;                      // 17 uint4 per LDS row
    const uint4* WtQ = (const uint4*)Wt;

    // Hoist B fragments for all 4 K-steps (N = n0+m fixed, K = hi*16..+15 contiguous)
    BF16Frag b[4];
    #pragma unroll
    for (int kk = 0; kk < 4; ++kk) {
        int kq = kk * 4;
        b[kk].q[0] = WtQ[(n0 + m) * 17 + kq + hi * 2];
        b[kk].q[1] = WtQ[(n0 + m) * 17 + kq + hi * 2 + 1];
    }
    float bs = bias ? bias[n0 + m] : 0.f;

    #pragma unroll
    for (int mt = 0; mt < 8; ++mt) {
        const int rbase = mt * 16 + m;                        // A row (lane's M) within strip
        v8f acc = {};
        #pragma unroll
        for (int kk = 0; kk < 4; ++kk) {
            int kq = kk * 4;
            BF16Frag a;                                        // K = hi*16..+15 contiguous
            a.q[0] = AsQ[rbase * 17 + kq + hi * 2];
            a.q[1] = AsQ[rbase * 17 + kq + hi * 2 + 1];
            acc = __builtin_amdgcn_wmma_f32_16x16x32_bf16(false, a.v, false, b[kk].v,
                                                          (short)0, acc, false, false);
        }
        const int grow0 = row0 + mt * 16;
        if (grow0 < N_NODES) {                                // tiles are all-or-nothing (50000 % 16 == 0)
            #pragma unroll
            for (int r = 0; r < 8; ++r)                       // D: VGPR r -> M = r + 8*hi, N = m
                D[(size_t)(grow0 + r + hi * 8) * DIM + n0 + m] = acc[r] + bs;
        }
    }
}

// ---------------------------------------------------------------------------
// Per-node attention scalars: as[n] = h[n].a_s ; ad[n] = h[n].a_d (wave/node)
// ---------------------------------------------------------------------------
__global__ void asad_kernel(const float* __restrict__ h, const float* __restrict__ a_s,
                            const float* __restrict__ a_d,
                            float* __restrict__ os, float* __restrict__ od) {
    int gid = blockIdx.x * 256 + threadIdx.x;
    int node = gid >> 5, l = gid & 31;
    if (node >= N_NODES) return;
    const float* hr = h + (size_t)node * DIM;
    float ss = 0.f, sd = 0.f;
    for (int c = l; c < DIM; c += 32) { float v = hr[c]; ss += v * a_s[c]; sd += v * a_d[c]; }
    for (int ofs = 16; ofs; ofs >>= 1) { ss += __shfl_down(ss, ofs, 32); sd += __shfl_down(sd, ofs, 32); }
    if (l == 0) { os[node] = ss; od[node] = sd; }
}

__device__ __forceinline__ void edge_ends(const int* ei, int e, int& s, int& d) {
    if (e < N_EDGES) { s = ei[e]; d = ei[N_EDGES + e]; } else { s = d = e - N_EDGES; }
}

// alpha = leaky(as[src]+ad[dst]+ae, 0.2); segment-max via monotone-uint atomicMax
__global__ void alpha_max_kernel(const int* __restrict__ ei, const float* __restrict__ asb,
                                 const float* __restrict__ adb, const float* __restrict__ aeL,
                                 float* __restrict__ alpha, unsigned* __restrict__ mbuf) {
    int e = blockIdx.x * 256 + threadIdx.x;
    if (e >= E2) return;
    int s, d; edge_ends(ei, e, s, d);
    float a = asb[s] + adb[d] + aeL[e];
    a = (a > 0.f) ? a : 0.2f * a;
    alpha[e] = a;
    unsigned u = __float_as_uint(a);
    u ^= (u & 0x80000000u) ? 0xFFFFFFFFu : 0x80000000u;   // order-preserving flip
    atomicMax(&mbuf[d], u);
}

__global__ void exp_sum_kernel(const int* __restrict__ ei, const float* __restrict__ alpha,
                               const unsigned* __restrict__ mbuf,
                               float* __restrict__ exb, float* __restrict__ den) {
    int e = blockIdx.x * 256 + threadIdx.x;
    if (e >= E2) return;
    int s, d; edge_ends(ei, e, s, d);
    unsigned u = mbuf[d];
    u ^= (u & 0x80000000u) ? 0x80000000u : 0xFFFFFFFFu;   // un-flip
    float ex = __expf(alpha[e] - __uint_as_float(u));
    exb[e] = ex;
    atomicAdd(&den[d], ex);
}

// Wave per edge: acc[dst] += (ex/den[dst]) * h[src]  (coalesced 512B row, f32 atomics)
__global__ void aggregate_kernel(const int* __restrict__ ei, const float* __restrict__ h,
                                 const float* __restrict__ exb, const float* __restrict__ den,
                                 float* __restrict__ acc) {
    int gid = blockIdx.x * 256 + threadIdx.x;
    int e = gid >> 5, l = gid & 31;
    if (e >= E2) return;
    int s, d; edge_ends(ei, e, s, d);
    float coef = exb[e] / den[d];
    float4 v = ((const float4*)(h + (size_t)s * DIM))[l];
    float* o = acc + (size_t)d * DIM + l * 4;
    atomicAdd(o + 0, coef * v.x); atomicAdd(o + 1, coef * v.y);
    atomicAdd(o + 2, coef * v.z); atomicAdd(o + 3, coef * v.w);
}

__global__ void bias_act_kernel(const float* __restrict__ acc, const float* __restrict__ b,
                                float* __restrict__ feat, int do_act) {
    int idx = blockIdx.x * 256 + threadIdx.x;
    if (idx >= N_NODES * DIM) return;
    float v = acc[idx] + b[idx & 127];
    if (do_act) v = (v > 0.f) ? v : 0.01f * v;
    feat[idx] = v;
}

__global__ void colsum_kernel(const float* __restrict__ feat, float* __restrict__ vnsum) {
    __shared__ float sh[256];
    int c = threadIdx.x & 127, p = threadIdx.x >> 7;
    float a = 0.f;
    for (int r = blockIdx.x * 2 + p; r < N_NODES; r += gridDim.x * 2)
        a += feat[(size_t)r * DIM + c];
    sh[threadIdx.x] = a;
    __syncthreads();
    if (p == 0) atomicAdd(&vnsum[c], sh[threadIdx.x] + sh[threadIdx.x + 128]);
}

// vn = relu-MLP chain on a single 128-vector (4 layers), one block of 128 thr.
__global__ void vn_mlp_kernel(const float* __restrict__ vnsum, const float* __restrict__ vn_w,
                              const float* w1, const float* b1, const float* w2, const float* b2,
                              const float* w3, const float* b3, const float* w4, const float* b4,
                              float* __restrict__ outp) {
    __shared__ float v[DIM];
    int j = threadIdx.x;
    v[j] = vnsum[j] + vn_w[j];
    __syncthreads();
    const float* ws[4] = {w1, w2, w3, w4};
    const float* bs[4] = {b1, b2, b3, b4};
    for (int L = 0; L < 4; ++L) {
        float a = bs[L][j];
        for (int k = 0; k < DIM; ++k) a += v[k] * ws[L][k * DIM + j];
        a = fmaxf(a, 0.f);
        __syncthreads();
        v[j] = a;
        __syncthreads();
    }
    outp[j] = v[j];
}

// ---------------------------------------------------------------------------
extern "C" void kernel_launch(void* const* d_in, const int* in_sizes, int n_in,
                              void* d_out, int out_size, void* d_ws, size_t ws_size,
                              hipStream_t stream) {
    (void)in_sizes; (void)n_in; (void)out_size; (void)ws_size;
    const float* x  = (const float*)d_in[0];
    const int*   ei = (const int*)d_in[1];
    const float* ea = (const float*)d_in[2];
    const float* Wl[3]  = {(const float*)d_in[3], (const float*)d_in[9],  (const float*)d_in[15]};
    const float* asv[3] = {(const float*)d_in[4], (const float*)d_in[10], (const float*)d_in[16]};
    const float* adv[3] = {(const float*)d_in[5], (const float*)d_in[11], (const float*)d_in[17]};
    const float* Wev[3] = {(const float*)d_in[6], (const float*)d_in[12], (const float*)d_in[18]};
    const float* aev[3] = {(const float*)d_in[7], (const float*)d_in[13], (const float*)d_in[19]};
    const float* bv[3]  = {(const float*)d_in[8], (const float*)d_in[14], (const float*)d_in[20]};
    const float* vn_w = (const float*)d_in[21];
    const float* m1w1 = (const float*)d_in[22]; const float* m1b1 = (const float*)d_in[23];
    const float* m1w2 = (const float*)d_in[24]; const float* m1b2 = (const float*)d_in[25];
    const float* m2w1 = (const float*)d_in[26]; const float* m2b1 = (const float*)d_in[27];
    const float* m2w2 = (const float*)d_in[28]; const float* m2b2 = (const float*)d_in[29];
    const float* Wout = (const float*)d_in[30]; const float* bout = (const float*)d_in[31];
    float* outp = (float*)d_out;

    uintptr_t cur = (uintptr_t)d_ws;
    auto carve = [&](size_t bytes) -> void* {
        cur = (cur + 255) & ~(uintptr_t)255;
        void* p = (void*)cur; cur += bytes; return p;
    };
    float*    loop_attr = (float*)carve((size_t)N_NODES * EDIM * 4);
    float*    deg   = (float*)carve((size_t)N_NODES * 4);
    float*    wae   = (float*)carve(3 * 64 * 4);
    float*    ae3   = (float*)carve((size_t)3 * E2 * 4);
    float*    asb   = (float*)carve((size_t)N_NODES * 4);
    float*    adb   = (float*)carve((size_t)N_NODES * 4);
    float*    feat  = (float*)carve((size_t)N_NODES * DIM * 4);
    float*    h     = (float*)carve((size_t)N_NODES * DIM * 4);
    float*    acc   = (float*)carve((size_t)N_NODES * DIM * 4);
    float*    alpha = (float*)carve((size_t)E2 * 4);
    float*    exb   = (float*)carve((size_t)E2 * 4);
    unsigned* mbuf  = (unsigned*)carve((size_t)N_NODES * 4);
    float*    den   = (float*)carve((size_t)N_NODES * 4);
    float*    vnsum = (float*)carve(DIM * 4);

    hipMemsetAsync(deg, 0, (size_t)N_NODES * 4, stream);
    hipMemsetAsync(loop_attr, 0, (size_t)N_NODES * EDIM * 4, stream);
    deg_loop_kernel<<<(N_EDGES * EDIM + 255) / 256, 256, 0, stream>>>(ei, ea, deg, loop_attr);
    loop_norm_kernel<<<(N_NODES * EDIM + 255) / 256, 256, 0, stream>>>(deg, loop_attr);
    wae_kernel<<<1, 256, 0, stream>>>(Wev[0], aev[0], Wev[1], aev[1], Wev[2], aev[2], wae);
    edge_ae_kernel<<<(E2 * 16 + 255) / 256, 256, 0, stream>>>(ea, loop_attr, wae, ae3);

    const float* cur_in = x;
    for (int l = 0; l < 3; ++l) {
        gemm_wmma_kernel<<<GEMM_BLOCKS, 256, 0, stream>>>(cur_in, Wl[l], nullptr, h);
        asad_kernel<<<(N_NODES * 32 + 255) / 256, 256, 0, stream>>>(h, asv[l], adv[l], asb, adb);
        hipMemsetAsync(mbuf, 0, (size_t)N_NODES * 4, stream);
        hipMemsetAsync(den,  0, (size_t)N_NODES * 4, stream);
        hipMemsetAsync(acc,  0, (size_t)N_NODES * DIM * 4, stream);
        alpha_max_kernel<<<(E2 + 255) / 256, 256, 0, stream>>>(ei, asb, adb, ae3 + (size_t)l * E2, alpha, mbuf);
        exp_sum_kernel<<<(E2 + 255) / 256, 256, 0, stream>>>(ei, alpha, mbuf, exb, den);
        aggregate_kernel<<<(E2 * 32 + 255) / 256, 256, 0, stream>>>(ei, h, exb, den, acc);
        bias_act_kernel<<<(N_NODES * DIM + 255) / 256, 256, 0, stream>>>(acc, bv[l], feat, l > 0 ? 1 : 0);
        cur_in = feat;
    }

    gemm_wmma_kernel<<<GEMM_BLOCKS, 256, 0, stream>>>(feat, Wout, bout, outp);
    hipMemsetAsync(vnsum, 0, DIM * 4, stream);
    colsum_kernel<<<256, 256, 0, stream>>>(feat, vnsum);
    vn_mlp_kernel<<<1, 128, 0, stream>>>(vnsum, vn_w, m1w1, m1b1, m1w2, m1b2,
                                         m2w1, m2b1, m2w2, m2b2, outp + (size_t)N_NODES * DIM);
}
